// TreeTransformer_61649960567036
// MI455X (gfx1250) — matrix operand order repaired and checked by reading, not compiled, not run on hardware
//
#include <hip/hip_runtime.h>
#include <math.h>

typedef __attribute__((ext_vector_type(16))) __bf16 v16bf;
typedef __attribute__((ext_vector_type(8)))  float  v8f;
typedef __bf16 bf16;

#define DEVINL __device__ __forceinline__
#define NEG_INF (-__builtin_inff())

static constexpr int B_ = 16, T_ = 512, Hc = 256, Fc = 1024, Lc = 4, NHc = 8, Dh = 32, OUTc = 64;
static constexpr float EPSc = 1e-5f;

// ---------------- WMMA fragment loaders (bf16 16x16x32, wave32) ----------------
// A (16x32, MxK): lane = M (0..15), half=lane>>4; elem j -> K=16*half+j
DEVINL v16bf load_a_frag(const bf16* A, int lda, int lane) {
  int m = lane & 15, half = lane >> 4;
  const bf16* p = A + (size_t)m * lda + 16 * half;
  v16bf a;
#pragma unroll
  for (int j = 0; j < 16; ++j) a[j] = p[j];
  return a;
}

// B (32x16, KxN) built from row-major W[N][K] (computing X @ W^T):
// lane: n = lane&15, K = 16*(lane>>4) + j  ->  16 consecutive bf16 of W's row.
DEVINL v16bf load_b_frag(const bf16* Wt, int ldw, int lane) {
  int n = lane & 15, half = lane >> 4;
  const bf16* p = Wt + (size_t)n * ldw + 16 * half;
  v16bf b;
#pragma unroll
  for (int j = 0; j < 16; ++j) b[j] = p[j];
  return b;
}

// B (32x16, KxN) from a row-major KxN tile (V in attention): strided per-row gather.
DEVINL v16bf load_bkn_frag(const bf16* Vt, int ldv, int lane) {
  int n = lane & 15, half = lane >> 4;
  const bf16* p = Vt + (size_t)(16 * half) * ldv + n;
  v16bf b;
#pragma unroll
  for (int j = 0; j < 16; ++j) b[j] = p[(size_t)j * ldv];
  return b;
}

DEVINL v8f wmma_bf16(v16bf a, v16bf b, v8f c) {
  return __builtin_amdgcn_wmma_f32_16x16x32_bf16(false, a, false, b, (short)0, c, false, false);
}

// ---------------- elementwise / setup kernels ----------------
__global__ void cvt_bf16_kernel(const float* __restrict__ s, bf16* __restrict__ d, int n) {
  int i = blockIdx.x * blockDim.x + threadIdx.x;
  if (i < n) d[i] = (bf16)s[i];
}

__global__ void transpose_btc_kernel(const float* __restrict__ trees, float* __restrict__ x) {
  int idx = blockIdx.x * blockDim.x + threadIdx.x;
  if (idx >= B_ * T_ * Hc) return;
  int c = idx % Hc;
  int t = (idx / Hc) % T_;
  int b = idx / (Hc * T_);
  x[idx] = trees[((size_t)b * Hc + c) * T_ + t];
}

__global__ void mask_init_kernel(float* __restrict__ mask) {
  int bi = blockIdx.x;                 // b*T + i
  int i = bi % T_;
  size_t base = (size_t)bi * T_;
  for (int j = threadIdx.x; j < T_; j += blockDim.x)
    mask[base + j] = (j == i) ? 0.f : NEG_INF;
}

__global__ void mask_edges_kernel(const int* __restrict__ idxes, float* __restrict__ mask) {
  int idx = blockIdx.x * blockDim.x + threadIdx.x;
  const int n = T_ - 1;
  if (idx >= B_ * n * 3) return;
  int b = idx / (n * 3);
  int rem = idx % (n * 3);
  int i = rem / 3, c = rem % 3;
  int t = idxes[(size_t)b * 3 * n + (size_t)i * 3 + c];
  int pos = i + 1;
  int rel = (t > 0) ? t : pos;         // rel<=0 redirects to diagonal (no-op)
  float* mb = mask + (size_t)b * T_ * T_;
  mb[(size_t)pos * T_ + rel] = 0.f;    // all racers write 0.0 -> benign
  mb[(size_t)rel * T_ + pos] = 0.f;
}

__global__ __launch_bounds__(256) void ln_fwd_kernel(const float* __restrict__ x,
                                                     const float* __restrict__ g,
                                                     const float* __restrict__ bb,
                                                     bf16* __restrict__ y) {
  __shared__ float red[256];
  int row = blockIdx.x, tid = threadIdx.x;
  float v = x[(size_t)row * Hc + tid];
  red[tid] = v;
  __syncthreads();
#pragma unroll
  for (int s = 128; s > 0; s >>= 1) {
    if (tid < s) red[tid] += red[tid + s];
    __syncthreads();
  }
  float mean = red[0] * (1.0f / Hc);
  __syncthreads();
  float d = v - mean;
  red[tid] = d * d;
  __syncthreads();
#pragma unroll
  for (int s = 128; s > 0; s >>= 1) {
    if (tid < s) red[tid] += red[tid + s];
    __syncthreads();
  }
  float var = red[0] * (1.0f / Hc);
  float out = d * rsqrtf(var + EPSc) * g[tid] + bb[tid];
  y[(size_t)row * Hc + tid] = (bf16)out;
}

// ---------------- WMMA GEMM: C[M,N] = A[M,K](bf16) @ W[N,K]^T + bias, epilogue ----
// ep==0: store bf16; ep==1: += res (f32), store f32; ep==2: exact GELU, store bf16.
// One wave computes a 16x64 strip. Software-pipelined: next K-step's A + 4 B
// fragments are issued before the current 4 WMMAs, so the vmem waits overlap
// with matrix-pipe work instead of gating each WMMA.
__global__ __launch_bounds__(32) void gemm_wmma_kernel(
    const bf16* __restrict__ A, const bf16* __restrict__ W,
    const float* __restrict__ bias, const float* __restrict__ res,
    float* __restrict__ outF, bf16* __restrict__ outB,
    int M, int N, int K, int ep) {
  int lane = threadIdx.x;
  int ntiles = N >> 6;
  int mt = blockIdx.x / ntiles;
  int nt = blockIdx.x % ntiles;
  int m0 = mt * 16, n0 = nt * 64;

  v8f acc[4] = {v8f{}, v8f{}, v8f{}, v8f{}};
  const bf16* Abase = A + (size_t)m0 * K;
  const bf16* Wb[4];
#pragma unroll
  for (int t = 0; t < 4; ++t) Wb[t] = W + (size_t)(n0 + 16 * t) * K;

  // prologue
  v16bf a = load_a_frag(Abase, K, lane);
  v16bf b0 = load_b_frag(Wb[0], K, lane);
  v16bf b1 = load_b_frag(Wb[1], K, lane);
  v16bf b2 = load_b_frag(Wb[2], K, lane);
  v16bf b3 = load_b_frag(Wb[3], K, lane);

  for (int k0 = 0; k0 < K - 32; k0 += 32) {
    int kn = k0 + 32;
    v16bf an  = load_a_frag(Abase + kn, K, lane);
    v16bf bn0 = load_b_frag(Wb[0] + kn, K, lane);
    v16bf bn1 = load_b_frag(Wb[1] + kn, K, lane);
    v16bf bn2 = load_b_frag(Wb[2] + kn, K, lane);
    v16bf bn3 = load_b_frag(Wb[3] + kn, K, lane);
    acc[0] = wmma_bf16(a, b0, acc[0]);
    acc[1] = wmma_bf16(a, b1, acc[1]);
    acc[2] = wmma_bf16(a, b2, acc[2]);
    acc[3] = wmma_bf16(a, b3, acc[3]);
    a = an; b0 = bn0; b1 = bn1; b2 = bn2; b3 = bn3;
  }
  acc[0] = wmma_bf16(a, b0, acc[0]);
  acc[1] = wmma_bf16(a, b1, acc[1]);
  acc[2] = wmma_bf16(a, b2, acc[2]);
  acc[3] = wmma_bf16(a, b3, acc[3]);

  int nid = lane & 15, half = lane >> 4;
#pragma unroll
  for (int t = 0; t < 4; ++t) {
    int n = n0 + t * 16 + nid;
    float bv = bias ? bias[n] : 0.f;
#pragma unroll
    for (int r = 0; r < 8; ++r) {
      int m = m0 + r + 8 * half;
      size_t idx = (size_t)m * N + n;
      float v = acc[t][r] + bv;
      if (ep == 1) {
        v += res[idx];
        outF[idx] = v;
      } else if (ep == 2) {
        v = 0.5f * v * (1.0f + erff(v * 0.70710678118654752f));
        outB[idx] = (bf16)v;
      } else {
        outB[idx] = (bf16)v;
      }
    }
  }
}

// ---------------- Flash attention, one wave per (b, h, 16-row tile) ----------------
__global__ __launch_bounds__(32) void attn_wmma_kernel(
    const bf16* __restrict__ Q, const bf16* __restrict__ Kk,
    const bf16* __restrict__ V, const float* __restrict__ mask,
    bf16* __restrict__ O) {
  __shared__ bf16 Pl[16 * 32];
  int lane = threadIdx.x;
  int nid = lane & 15, half = lane >> 4;
  int rt = blockIdx.x & 31;            // T/16 = 32 row tiles
  int h  = (blockIdx.x >> 5) & 7;
  int b  = blockIdx.x >> 8;
  int m0 = rt * 16;

  const bf16* qh = Q + (size_t)b * T_ * Hc + h * Dh;
  const bf16* kh = Kk + (size_t)b * T_ * Hc + h * Dh;
  const bf16* vh = V + (size_t)b * T_ * Hc + h * Dh;
  const float* mb = mask + (size_t)b * T_ * T_;
  const float scale = 0.17677669529663687f;  // 1/sqrt(32)

  v16bf qa = load_a_frag(qh + (size_t)m0 * Hc, Hc, lane);  // Q tile 16 x 32(d)
  v8f o0 = {}, o1 = {};
  float mx[8], ls[8];
#pragma unroll
  for (int r = 0; r < 8; ++r) { mx[r] = -1e30f; ls[r] = 0.f; }

  for (int s0 = 0; s0 < T_; s0 += 32) {
    // scores 16x32 = Q(16x32d) @ K^T : two K=32 WMMAs
    v16bf kb0 = load_b_frag(kh + (size_t)s0 * Hc, Hc, lane);
    v16bf kb1 = load_b_frag(kh + (size_t)(s0 + 16) * Hc, Hc, lane);
    v8f z = {};
    v8f st0 = wmma_bf16(qa, kb0, z);
    v8f st1 = wmma_bf16(qa, kb1, z);

    // issue V-tile loads now; they complete while we do softmax + LDS bounce
    v16bf vb0 = load_bkn_frag(vh + (size_t)s0 * Hc, Hc, lane);
    v16bf vb1 = load_bkn_frag(vh + (size_t)s0 * Hc + 16, Hc, lane);

    float p0[8], p1[8];
#pragma unroll
    for (int r = 0; r < 8; ++r) {
      int row = m0 + r + 8 * half;
      float s0v = st0[r] * scale + mb[(size_t)row * T_ + s0 + nid];
      float s1v = st1[r] * scale + mb[(size_t)row * T_ + s0 + 16 + nid];
      float rm = fmaxf(s0v, s1v);
#pragma unroll
      for (int off = 1; off < 16; off <<= 1) rm = fmaxf(rm, __shfl_xor(rm, off, 16));
      float nm = fmaxf(mx[r], rm);           // >= -1e30 -> always finite
      float corr = __expf(mx[r] - nm);
      mx[r] = nm;
      float e0 = __expf(s0v - nm);           // -inf score -> exp(-inf)=0, no NaN
      float e1 = __expf(s1v - nm);
      float rs = e0 + e1;
#pragma unroll
      for (int off = 1; off < 16; off <<= 1) rs += __shfl_xor(rs, off, 16);
      ls[r] = ls[r] * corr + rs;
      o0[r] *= corr;
      o1[r] *= corr;
      p0[r] = e0;
      p1[r] = e1;
    }

    // re-layout P (C-layout) -> A-fragment via LDS
    __syncthreads();
#pragma unroll
    for (int r = 0; r < 8; ++r) {
      Pl[(r + 8 * half) * 32 + nid] = (bf16)p0[r];
      Pl[(r + 8 * half) * 32 + 16 + nid] = (bf16)p1[r];
    }
    __syncthreads();
    v16bf pf = load_a_frag(Pl, 32, lane);

    // O(16x32d) += P(16x32s) @ V(32s x 32d) : two WMMAs
    o0 = wmma_bf16(pf, vb0, o0);
    o1 = wmma_bf16(pf, vb1, o1);
  }

  bf16* oh = O + (size_t)b * T_ * Hc + h * Dh;
#pragma unroll
  for (int r = 0; r < 8; ++r) {
    float inv = 1.f / ls[r];
    int row = m0 + r + 8 * half;
    oh[(size_t)row * Hc + nid] = (bf16)(o0[r] * inv);
    oh[(size_t)row * Hc + 16 + nid] = (bf16)(o1[r] * inv);
  }
}

// ---------------- final projection: only token 0 matters ----------------
__global__ __launch_bounds__(64) void out_proj_kernel(const float* __restrict__ x,
                                                      const float* __restrict__ W,
                                                      const float* __restrict__ bias,
                                                      float* __restrict__ out) {
  int b = blockIdx.x, o = threadIdx.x;
  const float* xr = x + (size_t)b * T_ * Hc;  // row t=0
  const float* wr = W + (size_t)o * Hc;
  float s = bias[o];
#pragma unroll 4
  for (int k = 0; k < Hc; ++k) s += xr[k] * wr[k];
  out[(size_t)b * OUTc + o] = s;
}

// ---------------- host ----------------
extern "C" void kernel_launch(void* const* d_in, const int* in_sizes, int n_in,
                              void* d_out, int out_size, void* d_ws, size_t ws_size,
                              hipStream_t stream) {
  (void)in_sizes; (void)n_in; (void)out_size; (void)ws_size;
  const float* trees = (const float*)d_in[0];
  const int*   idxes = (const int*)d_in[1];
  const float* ln1_g = (const float*)d_in[2];
  const float* ln1_b = (const float*)d_in[3];
  const float* qW = (const float*)d_in[4];   const float* qb = (const float*)d_in[5];
  const float* kW = (const float*)d_in[6];   const float* kb = (const float*)d_in[7];
  const float* vW = (const float*)d_in[8];   const float* vb = (const float*)d_in[9];
  const float* oW = (const float*)d_in[10];  const float* ob = (const float*)d_in[11];
  const float* ln2_g = (const float*)d_in[12];
  const float* ln2_b = (const float*)d_in[13];
  const float* f1W = (const float*)d_in[14]; const float* f1b = (const float*)d_in[15];
  const float* f2W = (const float*)d_in[16]; const float* f2b = (const float*)d_in[17];
  const float* outW = (const float*)d_in[18];
  const float* outb = (const float*)d_in[19];
  float* out = (float*)d_out;

  char* ws = (char*)d_ws;
  size_t off = 0;
  auto carve = [&](size_t bytes) -> char* {
    char* p = ws + off;
    off += (bytes + 255) & ~(size_t)255;
    return p;
  };
  const size_t MT = (size_t)B_ * T_;
  float* x    = (float*)carve(MT * Hc * 4);
  float* mask = (float*)carve((size_t)B_ * T_ * T_ * 4);
  bf16* ybf = (bf16*)carve(MT * Hc * 2);
  bf16* qbf = (bf16*)carve(MT * Hc * 2);
  bf16* kbf = (bf16*)carve(MT * Hc * 2);
  bf16* vbf = (bf16*)carve(MT * Hc * 2);
  bf16* abf = (bf16*)carve(MT * Hc * 2);
  bf16* hbf = (bf16*)carve(MT * Fc * 2);
  bf16* qWb = (bf16*)carve((size_t)Lc * Hc * Hc * 2);
  bf16* kWb = (bf16*)carve((size_t)Lc * Hc * Hc * 2);
  bf16* vWb = (bf16*)carve((size_t)Lc * Hc * Hc * 2);
  bf16* oWb = (bf16*)carve((size_t)Lc * Hc * Hc * 2);
  bf16* f1Wb = (bf16*)carve((size_t)Lc * Fc * Hc * 2);
  bf16* f2Wb = (bf16*)carve((size_t)Lc * Hc * Fc * 2);

  // weights -> bf16
  int nw = Lc * Hc * Hc;
  cvt_bf16_kernel<<<(nw + 255) / 256, 256, 0, stream>>>(qW, qWb, nw);
  cvt_bf16_kernel<<<(nw + 255) / 256, 256, 0, stream>>>(kW, kWb, nw);
  cvt_bf16_kernel<<<(nw + 255) / 256, 256, 0, stream>>>(vW, vWb, nw);
  cvt_bf16_kernel<<<(nw + 255) / 256, 256, 0, stream>>>(oW, oWb, nw);
  int nf = Lc * Fc * Hc;
  cvt_bf16_kernel<<<(nf + 255) / 256, 256, 0, stream>>>(f1W, f1Wb, nf);
  cvt_bf16_kernel<<<(nf + 255) / 256, 256, 0, stream>>>(f2W, f2Wb, nf);

  // x = trees^T, tree attention mask
  int nx = B_ * T_ * Hc;
  transpose_btc_kernel<<<(nx + 255) / 256, 256, 0, stream>>>(trees, x);
  mask_init_kernel<<<B_ * T_, 256, 0, stream>>>(mask);
  int ne = B_ * (T_ - 1) * 3;
  mask_edges_kernel<<<(ne + 255) / 256, 256, 0, stream>>>(idxes, mask);

  const int Mrows = B_ * T_;
  const int gH = (Mrows / 16) * (Hc / 64);
  const int gF = (Mrows / 16) * (Fc / 64);
  for (int l = 0; l < Lc; ++l) {
    ln_fwd_kernel<<<Mrows, 256, 0, stream>>>(x, ln1_g + l * Hc, ln1_b + l * Hc, ybf);
    gemm_wmma_kernel<<<gH, 32, 0, stream>>>(ybf, qWb + (size_t)l * Hc * Hc, qb + l * Hc,
                                            nullptr, nullptr, qbf, Mrows, Hc, Hc, 0);
    gemm_wmma_kernel<<<gH, 32, 0, stream>>>(ybf, kWb + (size_t)l * Hc * Hc, kb + l * Hc,
                                            nullptr, nullptr, kbf, Mrows, Hc, Hc, 0);
    gemm_wmma_kernel<<<gH, 32, 0, stream>>>(ybf, vWb + (size_t)l * Hc * Hc, vb + l * Hc,
                                            nullptr, nullptr, vbf, Mrows, Hc, Hc, 0);
    attn_wmma_kernel<<<B_ * NHc * (T_ / 16), 32, 0, stream>>>(qbf, kbf, vbf, mask, abf);
    gemm_wmma_kernel<<<gH, 32, 0, stream>>>(abf, oWb + (size_t)l * Hc * Hc, ob + l * Hc,
                                            x, x, nullptr, Mrows, Hc, Hc, 1);
    ln_fwd_kernel<<<Mrows, 256, 0, stream>>>(x, ln2_g + l * Hc, ln2_b + l * Hc, ybf);
    gemm_wmma_kernel<<<gF, 32, 0, stream>>>(ybf, f1Wb + (size_t)l * Fc * Hc, f1b + l * Fc,
                                            nullptr, nullptr, hbf, Mrows, Fc, Hc, 2);
    gemm_wmma_kernel<<<gH, 32, 0, stream>>>(hbf, f2Wb + (size_t)l * Hc * Fc, f2b + l * Hc,
                                            x, x, nullptr, Mrows, Hc, Fc, 1);
  }
  out_proj_kernel<<<B_, 64, 0, stream>>>(x, outW, outb, out);
}
